// STGformer2_28621662060947
// MI455X (gfx1250) — matrix operand order repaired and checked by reading, not compiled, not run on hardware
//
#include <hip/hip_runtime.h>

typedef _Float16 h16;
typedef __attribute__((ext_vector_type(16))) _Float16 v16h;
typedef __attribute__((ext_vector_type(8)))  float    v8f;

union Frag { v16h v; uint4 q[2]; };

// ---------------------------------------------------------------------------
// Generic batched WMMA GEMM:  C[m,n] = sum_k A[m,k]*B[n,k]  (+bias[n]) (relu)
// A: [M,K] row-major f16 (optionally A+A2 summed), B: [N,K] row-major f16.
// One wave computes MREP vertically-stacked 16x16 C tiles, sharing each B
// fragment across MREP v_wmma_f32_16x16x32_f16 issues.
// All M,N multiples of 16; all K multiples of 32; all ld* multiples of 8 halves.
// ---------------------------------------------------------------------------
template <int MREP, bool HASA2>
__global__ void k_gemm(const h16* __restrict__ A, const h16* __restrict__ A2,
                       const h16* __restrict__ B, const float* __restrict__ bias,
                       float* __restrict__ Cf, h16* __restrict__ Ch,
                       int Mg, int Nt, int Ki, int lda, int ldb, int ldc,
                       long long aS, long long a2S, long long bS, long long cS,
                       int aMod, int total, int relu) {
  int w = blockIdx.x * (blockDim.x >> 5) + (threadIdx.x >> 5);
  if (w >= total) return;                      // whole-wave uniform exit
  int lane = threadIdx.x & 31;
  int tpb  = Mg * Nt;
  int batch = w / tpb;
  int rem   = w - batch * tpb;
  int tg = rem / Nt;
  int tn = rem - tg * Nt;
  int hi = lane >> 4;        // K-half select
  int lm = lane & 15;        // row (A) / col (B,C)

  const h16* Ab  = A + (long long)(aMod ? (batch % aMod) : batch) * aS;
  const h16* Bb  = B + (long long)batch * bS;
  const h16* A2b = HASA2 ? (A2 + (long long)batch * a2S) : (const h16*)0;

  long long aRow[MREP];
#pragma unroll
  for (int r = 0; r < MREP; ++r)
    aRow[r] = (long long)(tg * (16 * MREP) + r * 16 + lm) * lda;
  long long bRow = (long long)(tn * 16 + lm) * ldb;

  v8f acc[MREP];
#pragma unroll
  for (int r = 0; r < MREP; ++r) acc[r] = (v8f){0.f,0.f,0.f,0.f,0.f,0.f,0.f,0.f};

  for (int kk = 0; kk < Ki; ++kk) {
    int k = kk * 32;
    // B fragment: lanes<16: K=[k..k+7],[k+16..k+23]; lanes>=16: +8
    Frag fb;
    const uint4* pb = (const uint4*)(Bb + bRow + k + 8 * hi);
    fb.q[0] = pb[0];
    fb.q[1] = pb[2];
#pragma unroll
    for (int r = 0; r < MREP; ++r) {
      // A fragment: lanes<16: K=[k..k+7],[k+16..k+23]; lanes>=16: +8
      Frag fa;
      fa.q[0] = *(const uint4*)(Ab + aRow[r] + k + 8 * hi);
      fa.q[1] = *(const uint4*)(Ab + aRow[r] + k + 16 + 8 * hi);
      if (HASA2) {
        Frag f2;
        f2.q[0] = *(const uint4*)(A2b + aRow[r] + k + 8 * hi);
        f2.q[1] = *(const uint4*)(A2b + aRow[r] + k + 16 + 8 * hi);
        fa.v = fa.v + f2.v;                    // fuse (G + P) operand
      }
      acc[r] = __builtin_amdgcn_wmma_f32_16x16x32_f16(false, fa.v, false, fb.v,
                                                      (short)0, acc[r], false, false);
    }
  }
  int nn = tn * 16 + lm;
  float bv = bias ? bias[nn] : 0.0f;
  long long cb = (long long)batch * cS;
#pragma unroll
  for (int r = 0; r < MREP; ++r) {
    int rbase = tg * (16 * MREP) + r * 16 + 8 * hi;   // C layout: M = v + 8*hi
#pragma unroll
    for (int v = 0; v < 8; ++v) {
      float xv = acc[r][v] + bv;
      if (relu && xv < 0.f) xv = 0.f;
      long long idx = cb + (long long)(rbase + v) * ldc + nn;
      if (Cf) Cf[idx] = xv;
      if (Ch) Ch[idx] = (h16)xv;
    }
  }
}

// ---------------------------------------------------------------------------
__global__ void k_cast(const float* __restrict__ s, h16* __restrict__ d, int n) {
  int i = blockIdx.x * blockDim.x + threadIdx.x;
  if (i < n) d[i] = (h16)s[i];
}

__global__ void k_zero(float* __restrict__ p, int n) {
  int i = blockIdx.x * blockDim.x + threadIdx.x;
  if (i < n) p[i] = 0.0f;
}

// embedding: h0 = [x@fc1+b | day_emb | week_emb | adp]   (f16, [B*T*N,160])
__global__ void k_embed(const float* __restrict__ x, const float* __restrict__ w1,
                        const float* __restrict__ b1, const float* __restrict__ dayE,
                        const float* __restrict__ weekE, const float* __restrict__ adp,
                        h16* __restrict__ h0, int total) {
  int i = blockIdx.x * blockDim.x + threadIdx.x;
  if (i >= total) return;
  int row = i / 160, col = i - row * 160;
  float out;
  if (col < 64) {
    const float* xr = x + (size_t)row * 3;
    out = b1[col] + xr[0] * w1[col * 3] + xr[1] * w1[col * 3 + 1] + xr[2] * w1[col * 3 + 2];
  } else if (col < 96) {
    int di = (int)(x[(size_t)row * 3 + 1] * 288.0f);
    di = di < 0 ? 0 : (di > 287 ? 287 : di);
    out = dayE[di * 32 + (col - 64)];
  } else if (col < 128) {
    int wi = (int)x[(size_t)row * 3 + 2];
    wi = wi < 0 ? 0 : (wi > 6 ? 6 : wi);
    out = weekE[wi * 32 + (col - 96)];
  } else {
    int t = (row >> 10) % 12;                  // row = (b*12+t)*1024+n
    int n = row & 1023;
    out = adp[((size_t)(t * 1024 + n)) * 32 + (col - 128)];
  }
  h0[i] = (h16)out;
}

// batched f16 transpose [R,C] -> [C,R], block (32,8), LDS tiled
__global__ void k_tr16(const h16* __restrict__ in, h16* __restrict__ out, int R, int C) {
  __shared__ h16 t[32][33];
  long long bo = (long long)blockIdx.z * R * C;
  int x = blockIdx.x * 32 + threadIdx.x;
  for (int j = 0; j < 32; j += 8) {
    int y = blockIdx.y * 32 + threadIdx.y + j;
    if (x < C && y < R) t[threadIdx.y + j][threadIdx.x] = in[bo + (long long)y * C + x];
  }
  __syncthreads();
  int x2 = blockIdx.y * 32 + threadIdx.x;
  for (int j = 0; j < 32; j += 8) {
    int y2 = blockIdx.x * 32 + threadIdx.y + j;
    if (x2 < R && y2 < C) out[bo + (long long)y2 * R + x2] = t[threadIdx.x][threadIdx.y + j];
  }
}

// l2-normalize (or just cast) 160-wide rows f32 -> f16, one wave per row
__global__ void k_rownorm(const float* __restrict__ src, h16* __restrict__ dst,
                          int rows, int doNorm) {
  int row = blockIdx.x * 8 + (threadIdx.x >> 5);
  if (row >= rows) return;
  int lane = threadIdx.x & 31;
  const float* s = src + (size_t)row * 160;
  float v[5]; float ss = 0.f;
  for (int i = 0; i < 5; ++i) { v[i] = s[lane + 32 * i]; ss += v[i] * v[i]; }
  for (int m = 16; m >= 1; m >>= 1) ss += __shfl_xor(ss, m, 32);
  float inv = 1.0f;
  if (doNorm) inv = 1.0f / fmaxf(sqrtf(ss), 1e-12f);
  h16* d = dst + (size_t)row * 160;
  for (int i = 0; i < 5; ++i) d[lane + 32 * i] = (h16)(v[i] * inv);
}

// partial Ksum: grid (4, 48); block sums 256 rows, atomic into ks[b*160+d]
__global__ void k_ksum(const h16* __restrict__ Kn, float* __restrict__ ks) {
  int b = blockIdx.x; int d = threadIdx.x;
  if (d >= 160) return;
  int r0 = blockIdx.y * 256;
  const h16* p = Kn + ((size_t)b * 12288 + r0) * 160 + d;
  float s = 0.f;
  for (int r = 0; r < 256; ++r) s += (float)p[(size_t)r * 160];
  atomicAdd(&ks[b * 160 + d], s);
}

// Dden[row] = Qn_row . Ksum[b] + 12
__global__ void k_dden(const h16* __restrict__ Qn, const float* __restrict__ ks,
                       float* __restrict__ dd, int rows) {
  int row = blockIdx.x * 8 + (threadIdx.x >> 5);
  if (row >= rows) return;
  int lane = threadIdx.x & 31;
  int b = row / 12288;
  const h16* q = Qn + (size_t)row * 160;
  const float* k = ks + b * 160;
  float s = 0.f;
  for (int i = 0; i < 5; ++i) { int c = lane + 32 * i; s += (float)q[c] * k[c]; }
  for (int m = 16; m >= 1; m >>= 1) s += __shfl_xor(s, m, 32);
  if (lane == 0) dd[row] = s + 12.0f;
}

// in-place softmax(relu(.)) over 1024-wide f16 rows, block=256
__global__ void k_softmax(h16* __restrict__ G) {
  __shared__ float red[8];
  int tid = threadIdx.x;
  h16* g = G + (size_t)blockIdx.x * 1024;
  float v[4]; float mx = 0.0f;
  for (int i = 0; i < 4; ++i) {
    float x = fmaxf((float)g[tid + 256 * i], 0.0f);
    v[i] = x; mx = fmaxf(mx, x);
  }
  for (int m = 16; m >= 1; m >>= 1) mx = fmaxf(mx, __shfl_xor(mx, m, 32));
  if ((tid & 31) == 0) red[tid >> 5] = mx;
  __syncthreads();
  if (tid == 0) { float t = red[0]; for (int i = 1; i < 8; ++i) t = fmaxf(t, red[i]); red[0] = t; }
  __syncthreads();
  mx = red[0];
  __syncthreads();
  float s = 0.f;
  for (int i = 0; i < 4; ++i) { v[i] = __expf(v[i] - mx); s += v[i]; }
  for (int m = 16; m >= 1; m >>= 1) s += __shfl_xor(s, m, 32);
  if ((tid & 31) == 0) red[tid >> 5] = s;
  __syncthreads();
  if (tid == 0) { float t = 0.f; for (int i = 0; i < 8; ++i) t += red[i]; red[0] = t; }
  __syncthreads();
  float inv = 1.0f / red[0];
  for (int i = 0; i < 4; ++i) g[tid + 256 * i] = (h16)(v[i] * inv);
}

// row & column sums of A [1024,1024]
__global__ void k_asums(const float* __restrict__ A, float* __restrict__ rs,
                        float* __restrict__ cs) {
  int j = blockIdx.x * blockDim.x + threadIdx.x;
  if (j >= 1024) return;
  float c = 0.f;
  for (int i = 0; i < 1024; ++i) c += A[(size_t)i * 1024 + j];
  float r = 0.f;
  const float* ar = A + (size_t)j * 1024;
  for (int k = 0; k < 1024; ++k) r += ar[k];
  rs[j] = r; cs[j] = c;
}

// P[i,j] = A[i,j]/rowsum[j] + A[j,i]/colsum[j]  (torch broadcast semantics)
__global__ void k_pbuild(const float* __restrict__ A, const float* __restrict__ rs,
                         const float* __restrict__ cs, h16* __restrict__ P) {
  int i = blockIdx.x * blockDim.x + threadIdx.x;
  if (i >= 1024 * 1024) return;
  int r = i >> 10, c = i & 1023;
  P[i] = (h16)(A[i] / rs[c] + A[(size_t)c * 1024 + r] / cs[c]);
}

// CAT halves: (raw + 12*V) / (Dden * 1024), write f32 + f16
__global__ void k_combine(float* __restrict__ CATf, h16* __restrict__ CATh,
                          const h16* __restrict__ V, const float* __restrict__ dd,
                          int total) {
  int i = blockIdx.x * blockDim.x + threadIdx.x;
  if (i >= total) return;
  int row = i / 160, d = i - row * 160;
  float inv = 1.0f / (dd[row] * 1024.0f);
  float vv = 12.0f * (float)V[i];
  size_t o = (size_t)row * 320 + d;
  float a = (CATf[o] + vv) * inv;
  float b = (CATf[o + 160] + vv) * inv;
  CATf[o] = a;       CATf[o + 160] = b;
  CATh[o] = (h16)a;  CATh[o + 160] = (h16)b;
}

// out = base + att*pw*scale   (out may alias base)
__global__ void k_xglo(float* __restrict__ out, const float* __restrict__ base,
                       const float* __restrict__ att, const float* __restrict__ pw,
                       float scale, int n) {
  int i = blockIdx.x * blockDim.x + threadIdx.x;
  if (i < n) out[i] = base[i] + att[i] * pw[i] * scale;
}

// LayerNorm over 160: in = alpha*X (+Y); one wave per row
__global__ void k_ln(const float* __restrict__ X, const float* __restrict__ Y,
                     float alpha, const float* __restrict__ g, const float* __restrict__ b,
                     float* __restrict__ outF, h16* __restrict__ outH, int rows) {
  int row = blockIdx.x * 8 + (threadIdx.x >> 5);
  if (row >= rows) return;
  int lane = threadIdx.x & 31;
  size_t base = (size_t)row * 160;
  float v[5]; float s = 0.f;
  for (int i = 0; i < 5; ++i) {
    int c = lane + 32 * i;
    float x = alpha * X[base + c];
    if (Y) x += Y[base + c];
    v[i] = x; s += x;
  }
  for (int m = 16; m >= 1; m >>= 1) s += __shfl_xor(s, m, 32);
  float mu = s / 160.f;
  float var = 0.f;
  for (int i = 0; i < 5; ++i) { float d = v[i] - mu; var += d * d; }
  for (int m = 16; m >= 1; m >>= 1) var += __shfl_xor(var, m, 32);
  float rstd = rsqrtf(var / 160.f + 1e-5f);
  for (int i = 0; i < 5; ++i) {
    int c = lane + 32 * i;
    float y = (v[i] - mu) * rstd * g[c] + b[c];
    if (outF) outF[base + c] = y;
    if (outH) outH[base + c] = (h16)y;
  }
}

__global__ void k_addcast(float* __restrict__ z, const float* __restrict__ t,
                          h16* __restrict__ zh, int n) {
  int i = blockIdx.x * blockDim.x + threadIdx.x;
  if (i < n) { float v = z[i] + t[i]; z[i] = v; zh[i] = (h16)v; }
}

// pad fc2 [12,160]->[16,160] f16 and bias [12]->[16] f32
__global__ void k_padfc2(const float* __restrict__ wsrc, const float* __restrict__ bsrc,
                         h16* __restrict__ wdst, float* __restrict__ bdst) {
  int i = blockIdx.x * blockDim.x + threadIdx.x;
  if (i < 16 * 160) {
    int r = i / 160, c = i - r * 160;
    wdst[i] = (h16)(r < 12 ? wsrc[r * 160 + c] : 0.0f);
  } else if (i < 16 * 160 + 16) {
    int r = i - 16 * 160;
    bdst[r] = r < 12 ? bsrc[r] : 0.0f;
  }
}

// hsa [b,t,n,d] f32 -> zin [b,n,t*160+d] f16
__global__ void k_trhsa(const float* __restrict__ hsa, h16* __restrict__ zin, int total) {
  int i = blockIdx.x * blockDim.x + threadIdx.x;
  if (i >= total) return;
  int d = i % 160; int r = i / 160;
  int n = r & 1023; int bt = r >> 10; int t = bt % 12; int b = bt / 12;
  zin[((((size_t)b * 1024 + n) * 12) + t) * 160 + d] = (h16)hsa[i];
}

// zo [b,n,16] -> out [b,s,n]
__global__ void k_out(const float* __restrict__ zo, float* __restrict__ out) {
  int i = blockIdx.x * blockDim.x + threadIdx.x;
  if (i >= 4 * 12 * 1024) return;
  int n = i & 1023; int bs = i >> 10; int s = bs % 12; int b = bs / 12;
  out[i] = zo[((size_t)b * 1024 + n) * 16 + s];
}

// ===========================================================================
extern "C" void kernel_launch(void* const* d_in, const int* in_sizes, int n_in,
                              void* d_out, int out_size, void* d_ws, size_t ws_size,
                              hipStream_t stream) {
  (void)in_sizes; (void)n_in; (void)out_size; (void)ws_size;
  const int M = 49152, Mz = 4096;

  const float* x     = (const float*)d_in[0];
  const float* Amat  = (const float*)d_in[1];
  const float* fc1w  = (const float*)d_in[2];
  const float* fc1b  = (const float*)d_in[3];
  const float* dayE  = (const float*)d_in[4];
  const float* weekE = (const float*)d_in[5];
  const float* adp   = (const float*)d_in[6];
  const float* convw = (const float*)d_in[7];
  const float* convb = (const float*)d_in[8];
  const float* WQ[2]  = {(const float*)d_in[9],  (const float*)d_in[14]};
  const float* WK[2]  = {(const float*)d_in[10], (const float*)d_in[15]};
  const float* WV[2]  = {(const float*)d_in[11], (const float*)d_in[16]};
  const float* fcw[2] = {(const float*)d_in[12], (const float*)d_in[17]};
  const float* fcb[2] = {(const float*)d_in[13], (const float*)d_in[18]};
  const float* pww[2] = {(const float*)d_in[19], (const float*)d_in[21]};
  const float* pwb[2] = {(const float*)d_in[20], (const float*)d_in[22]};
  const float* l1w = (const float*)d_in[23]; const float* l1b = (const float*)d_in[24];
  const float* l2w = (const float*)d_in[25]; const float* l2b = (const float*)d_in[26];
  const float* ln1g = (const float*)d_in[27]; const float* ln1bv = (const float*)d_in[28];
  const float* ln2g = (const float*)d_in[29]; const float* ln2bv = (const float*)d_in[30];
  const float* e1w = (const float*)d_in[31]; const float* e1b = (const float*)d_in[32];
  const float* e2w[3][2]; const float* e2b[3][2];
  for (int r = 0; r < 3; ++r) {
    e2w[r][0] = (const float*)d_in[33 + 4 * r]; e2b[r][0] = (const float*)d_in[34 + 4 * r];
    e2w[r][1] = (const float*)d_in[35 + 4 * r]; e2b[r][1] = (const float*)d_in[36 + 4 * r];
  }
  const float* fc2w = (const float*)d_in[45]; const float* fc2b = (const float*)d_in[46];

  // ---- workspace arena (~640 MB) ----
  char* wsb = (char*)d_ws; size_t off = 0;
  auto alloc = [&](size_t bytes) -> void* {
    off = (off + 255) & ~(size_t)255;
    void* p = wsb + off; off += bytes; return p;
  };
  auto af = [&](size_t n) { return (float*)alloc(n * 4); };
  auto ah = [&](size_t n) { return (h16*)alloc(n * 2); };

  float* hF   = af((size_t)M * 160);
  float* qkvF = af((size_t)M * 160);
  float* CATf = af((size_t)M * 320);
  float* attF = af((size_t)M * 160);
  float* pwF  = af((size_t)M * 160);
  float* xgF  = af((size_t)M * 160);
  float* x1F  = af((size_t)M * 160);
  float* mloF = af((size_t)M * 160);
  float* hsaF = af((size_t)M * 160);
  float* ddF  = af(M);
  float* ksF  = af(4 * 160);
  float* rsF  = af(1024);
  float* csF  = af(1024);
  float* zF   = af((size_t)Mz * 160);
  float* ztF  = af((size_t)Mz * 160);
  float* zoF  = af((size_t)Mz * 16);
  float* fc2bp = af(16);

  h16* h0H  = ah((size_t)M * 160);
  h16* hH   = ah((size_t)M * 160);
  h16* hTH  = ah((size_t)M * 160);
  h16* GH   = ah((size_t)12 * 1024 * 1024);
  h16* PH   = ah((size_t)1024 * 1024);
  h16* xg1H = ah((size_t)M * 160);
  h16* QnH  = ah((size_t)M * 160);
  h16* KnH  = ah((size_t)M * 160);
  h16* VH   = ah((size_t)M * 160);
  h16* QnT  = ah((size_t)M * 160);
  h16* KnT  = ah((size_t)M * 160);
  h16* VT   = ah((size_t)M * 160);
  h16* M1tH = ah((size_t)48 * 160 * 160);
  h16* QtVtH= ah((size_t)48 * 160 * 160);
  h16* att0H= ah((size_t)M * 160);
  h16* CATh = ah((size_t)M * 320);
  h16* x1H  = ah((size_t)M * 160);
  h16* hidH = ah((size_t)M * 320);
  h16* zinH = ah((size_t)Mz * 1920);
  h16* zH   = ah((size_t)Mz * 160);
  h16* zhH  = ah((size_t)Mz * 320);
  h16* convH = ah(25600);
  h16* WQH[2] = {ah(25600), ah(25600)};
  h16* WKH[2] = {ah(25600), ah(25600)};
  h16* WVH[2] = {ah(25600), ah(25600)};
  h16* fcwH[2] = {ah(51200), ah(51200)};
  h16* pwH[2] = {ah(25600), ah(25600)};
  h16* l1H = ah(51200);
  h16* l2H = ah(51200);
  h16* e1H = ah(307200);
  h16* e2H[3][2];
  for (int r = 0; r < 3; ++r) { e2H[r][0] = ah(51200); e2H[r][1] = ah(51200); }
  h16* fc2H = ah(2560);
  h16* adpH = ah(393216);

  auto cast = [&](const float* s, h16* d, int n) {
    k_cast<<<(n + 255) / 256, 256, 0, stream>>>(s, d, n);
  };
  // dispatch over MREP (B-fragment reuse factor) and HASA2
  auto gemm = [&](const h16* A, const h16* A2, const h16* B, const float* bias,
                  float* Cf, h16* Ch, int Mr, int Nc, int K,
                  int lda, int ldb, int ldc,
                  long long aS, long long a2S, long long bS, long long cS,
                  int aMod, int batches, int relu) {
    int Mt = Mr / 16, Nt = Nc / 16, Ki = K / 32;
    int mrep = (Mt % 4 == 0) ? 4 : ((Mt % 2 == 0) ? 2 : 1);
    int Mg = Mt / mrep;
    int total = batches * Mg * Nt;
    dim3 grid((total + 3) / 4), blk(128);
    if (A2) {
      if (mrep == 4)      k_gemm<4,true><<<grid,blk,0,stream>>>(A,A2,B,bias,Cf,Ch,Mg,Nt,Ki,lda,ldb,ldc,aS,a2S,bS,cS,aMod,total,relu);
      else if (mrep == 2) k_gemm<2,true><<<grid,blk,0,stream>>>(A,A2,B,bias,Cf,Ch,Mg,Nt,Ki,lda,ldb,ldc,aS,a2S,bS,cS,aMod,total,relu);
      else                k_gemm<1,true><<<grid,blk,0,stream>>>(A,A2,B,bias,Cf,Ch,Mg,Nt,Ki,lda,ldb,ldc,aS,a2S,bS,cS,aMod,total,relu);
    } else {
      if (mrep == 4)      k_gemm<4,false><<<grid,blk,0,stream>>>(A,A2,B,bias,Cf,Ch,Mg,Nt,Ki,lda,ldb,ldc,aS,a2S,bS,cS,aMod,total,relu);
      else if (mrep == 2) k_gemm<2,false><<<grid,blk,0,stream>>>(A,A2,B,bias,Cf,Ch,Mg,Nt,Ki,lda,ldb,ldc,aS,a2S,bS,cS,aMod,total,relu);
      else                k_gemm<1,false><<<grid,blk,0,stream>>>(A,A2,B,bias,Cf,Ch,Mg,Nt,Ki,lda,ldb,ldc,aS,a2S,bS,cS,aMod,total,relu);
    }
  };

  // ---- weight casts ----
  cast(convw, convH, 25600);
  for (int li = 0; li < 2; ++li) {
    cast(WQ[li], WQH[li], 25600); cast(WK[li], WKH[li], 25600);
    cast(WV[li], WVH[li], 25600); cast(fcw[li], fcwH[li], 51200);
    cast(pww[li], pwH[li], 25600);
  }
  cast(l1w, l1H, 51200); cast(l2w, l2H, 51200); cast(e1w, e1H, 307200);
  for (int r = 0; r < 3; ++r) { cast(e2w[r][0], e2H[r][0], 51200); cast(e2w[r][1], e2H[r][1], 51200); }
  cast(adp, adpH, 393216);
  k_padfc2<<<(2576 + 255) / 256, 256, 0, stream>>>(fc2w, fc2b, fc2H, fc2bp);

  // ---- embed + conv ----
  k_embed<<<(M * 160 + 255) / 256, 256, 0, stream>>>(x, fc1w, fc1b, dayE, weekE, adp, h0H, M * 160);
  gemm(h0H, 0, convH, convb, hF, hH, M, 160, 160, 160, 160, 160, 0, 0, 0, 0, 0, 1, 0);
  k_tr16<<<dim3(5, 32, 48), dim3(32, 8), 0, stream>>>(hH, hTH, 1024, 160);

  // ---- graph matrices ----
  k_asums<<<4, 256, 0, stream>>>(Amat, rsF, csF);
  k_pbuild<<<(1024 * 1024 + 255) / 256, 256, 0, stream>>>(Amat, rsF, csF, PH);
  gemm(adpH, 0, adpH, 0, 0, GH, 1024, 1024, 32, 32, 32, 1024,
       32768, 0, 32768, 1048576, 0, 12, 0);
  k_softmax<<<12288, 256, 0, stream>>>(GH);
  // xg1 = (G[t] + P) @ h[b,t]   (A2 fused add, no exec-masking: templated)
  gemm(GH, PH, hTH, 0, 0, xg1H, 1024, 160, 1024, 1024, 1024, 160,
       1048576, 0, 163840, 163840, 12, 48, 0);

  // ---- attention layers ----
  auto attn_layer = [&](const h16* Xh, int li, h16* attH) {
    gemm(Xh, 0, WQH[li], 0, qkvF, 0, M, 160, 160, 160, 160, 160, 0, 0, 0, 0, 0, 1, 0);
    k_rownorm<<<M / 8, 256, 0, stream>>>(qkvF, QnH, M, 1);
    gemm(Xh, 0, WKH[li], 0, qkvF, 0, M, 160, 160, 160, 160, 160, 0, 0, 0, 0, 0, 1, 0);
    k_rownorm<<<M / 8, 256, 0, stream>>>(qkvF, KnH, M, 1);
    gemm(Xh, 0, WVH[li], 0, qkvF, 0, M, 160, 160, 160, 160, 160, 0, 0, 0, 0, 0, 1, 0);
    k_rownorm<<<M / 8, 256, 0, stream>>>(qkvF, VH, M, 0);
    k_tr16<<<dim3(5, 32, 48), dim3(32, 8), 0, stream>>>(QnH, QnT, 1024, 160);
    k_tr16<<<dim3(5, 32, 48), dim3(32, 8), 0, stream>>>(KnH, KnT, 1024, 160);
    k_tr16<<<dim3(5, 32, 48), dim3(32, 8), 0, stream>>>(VH,  VT,  1024, 160);
    k_zero<<<3, 256, 0, stream>>>(ksF, 640);
    k_ksum<<<dim3(4, 48), 160, 0, stream>>>(KnH, ksF);
    k_dden<<<M / 8, 256, 0, stream>>>(QnH, ksF, ddF, M);
    // rank-160 contractions: M1t = (Kn^T V)^T, QtVt = (Qn^T V)^T
    gemm(VT, 0, KnT, 0, 0, M1tH, 160, 160, 1024, 1024, 1024, 160,
         163840, 0, 163840, 25600, 0, 48, 0);
    gemm(VT, 0, QnT, 0, 0, QtVtH, 160, 160, 1024, 1024, 1024, 160,
         163840, 0, 163840, 25600, 0, 48, 0);
    // out_s -> CAT[:,0:160), out_t -> CAT[:,160:320)
    gemm(QnH, 0, M1tH, 0, CATf, 0, 1024, 160, 160, 160, 160, 320,
         163840, 0, 25600, 327680, 0, 48, 0);
    gemm(KnH, 0, QtVtH, 0, CATf + 160, 0, 1024, 160, 160, 160, 160, 320,
         163840, 0, 25600, 327680, 0, 48, 0);
    k_combine<<<(M * 160 + 255) / 256, 256, 0, stream>>>(CATf, CATh, VH, ddF, M * 160);
    gemm(CATh, 0, fcwH[li], fcb[li], attF, attH, M, 160, 320, 320, 320, 160,
         0, 0, 0, 0, 0, 1, 0);
  };

  attn_layer(hH, 0, att0H);
  gemm(hH, 0, pwH[0], pwb[0], pwF, 0, M, 160, 160, 160, 160, 160, 0, 0, 0, 0, 0, 1, 0);
  k_xglo<<<(M * 160 + 255) / 256, 256, 0, stream>>>(xgF, hF, attF, pwF, 1.0f, M * 160);

  attn_layer(xg1H, 1, 0);
  gemm(att0H, 0, pwH[1], pwb[1], pwF, 0, M, 160, 160, 160, 160, 160, 0, 0, 0, 0, 0, 1, 0);
  k_xglo<<<(M * 160 + 255) / 256, 256, 0, stream>>>(xgF, xgF, attF, pwF, 0.01f, M * 160);

  // ---- LN + MLP + LN ----
  k_ln<<<M / 8, 256, 0, stream>>>(xgF, 0, 2.0f, ln1g, ln1bv, x1F, x1H, M);
  gemm(x1H, 0, l1H, l1b, 0, hidH, M, 320, 160, 160, 160, 320, 0, 0, 0, 0, 0, 1, 1);
  gemm(hidH, 0, l2H, l2b, mloF, 0, M, 160, 320, 320, 320, 160, 0, 0, 0, 0, 0, 1, 0);
  k_ln<<<M / 8, 256, 0, stream>>>(x1F, mloF, 1.0f, ln2g, ln2bv, hsaF, 0, M);

  // ---- decode ----
  k_trhsa<<<(M * 160 + 255) / 256, 256, 0, stream>>>(hsaF, zinH, M * 160);
  gemm(zinH, 0, e1H, e1b, zF, zH, Mz, 160, 1920, 1920, 1920, 160, 0, 0, 0, 0, 0, 1, 0);
  for (int r = 0; r < 3; ++r) {
    gemm(zH, 0, e2H[r][0], e2b[r][0], 0, zhH, Mz, 320, 160, 160, 160, 320, 0, 0, 0, 0, 0, 1, 1);
    gemm(zhH, 0, e2H[r][1], e2b[r][1], ztF, 0, Mz, 160, 320, 320, 320, 160, 0, 0, 0, 0, 0, 1, 0);
    k_addcast<<<(Mz * 160 + 255) / 256, 256, 0, stream>>>(zF, ztF, zH, Mz * 160);
  }
  gemm(zH, 0, fc2H, fc2bp, zoF, 0, Mz, 16, 160, 160, 160, 16, 0, 0, 0, 0, 0, 1, 0);
  k_out<<<(49152 + 255) / 256, 256, 0, stream>>>(zoF, (float*)d_out);
}
